// QuantLinear_28973849379107
// MI455X (gfx1250) — matrix operand order repaired and checked by reading, not compiled, not run on hardware
//
#include <hip/hip_runtime.h>
#include <hip/hip_bf16.h>

typedef __attribute__((ext_vector_type(16))) _Float16 v16h;
typedef __attribute__((ext_vector_type(8)))  _Float16 v8h;
typedef __attribute__((ext_vector_type(2)))  _Float16 v2h;
typedef __attribute__((ext_vector_type(8)))  float    v8f;
typedef __attribute__((ext_vector_type(4)))  float    v4f;

#define BM 128
#define BN 128
#define BK 128            // == GPTQ group size -> one (scale, zero) group per K-step
#define LDS_STRIDE 136    // halves: 272B rows -> 16B-aligned, staggered banks

__global__ __launch_bounds__(256)
void gptq_wmma_kernel(const float* __restrict__ x,
                      const int*   __restrict__ qweight,
                      const float* __restrict__ scales,
                      const int*   __restrict__ qzeros,
                      float* __restrict__ out,
                      int M, int N, int K)
{
    __shared__ __align__(16) _Float16 As[BM * LDS_STRIDE];   // As[m][k]
    __shared__ __align__(16) _Float16 Bs[BN * LDS_STRIDE];   // Bs[n][k] (transposed)

    const int tid  = threadIdx.x;
    const int wave = tid >> 5;
    const int lane = tid & 31;
    const int hsel = lane >> 4;          // lane half of the wave
    const int lr   = lane & 15;

    const int wm = wave >> 1;            // 0..3 : wave row  (32 rows each)
    const int wn = wave & 1;             // 0..1 : wave col  (64 cols each)

    const int rowBlk = blockIdx.y * BM;
    const int colBlk = blockIdx.x * BN;
    const int Nw     = N >> 3;           // qzeros row width (words)

    // 8 independent accumulators: 2 (M tiles) x 4 (N tiles)
    v8f acc[2][4];
    #pragma unroll
    for (int i = 0; i < 2; ++i)
        #pragma unroll
        for (int j = 0; j < 4; ++j)
            acc[i][j] = (v8f){0.f,0.f,0.f,0.f,0.f,0.f,0.f,0.f};

    // B staging assignment: one column per (thread, K-half)
    const int nl = tid & 127;            // local column 0..127
    const int kh = tid >> 7;             // 0..1 : which 8 word-rows (64 k values)

    for (int kt = 0; kt < K; kt += BK) {
        const int g = kt >> 7;           // group index (GROUP = 128)
        __syncthreads();                 // previous iteration's LDS readers done

        // ---------- stage A: 128x128 fp32 -> f16 into LDS ----------
        // 2048 float8 chunks, 8 per thread; batch loads, then convert+store b128
        const float* Ablk = x + (size_t)rowBlk * K + kt;
        #pragma unroll
        for (int c = 0; c < 2; ++c) {
            v4f a0[4], a1[4];
            #pragma unroll
            for (int i = 0; i < 4; ++i) {
                int j  = tid + 256 * (c * 4 + i);    // 0..2047
                int r  = j >> 4;                     // 0..127
                int c8 = (j & 15) << 3;              // 0..120
                const float* p = Ablk + (size_t)r * K + c8;
                a0[i] = *(const v4f*)p;
                a1[i] = *(const v4f*)(p + 4);
            }
            #pragma unroll
            for (int i = 0; i < 4; ++i) {
                int j  = tid + 256 * (c * 4 + i);
                int r  = j >> 4;
                int c8 = (j & 15) << 3;
                v8h h;
                h[0] = (_Float16)a0[i].x; h[1] = (_Float16)a0[i].y;
                h[2] = (_Float16)a0[i].z; h[3] = (_Float16)a0[i].w;
                h[4] = (_Float16)a1[i].x; h[5] = (_Float16)a1[i].y;
                h[6] = (_Float16)a1[i].z; h[7] = (_Float16)a1[i].w;
                *(v8h*)&As[r * LDS_STRIDE + c8] = h;
            }
        }

        // ---------- stage B: packed 4-bit -> f16 dequant into LDS ----------
        // Magic-number trick: ((q | 0x6400) as f16) == 1024 + q, so
        // (q - z)*s == pk_fma(h2(1024+q), {s,s}, {-(1024+z)*s, ...})
        {
            const int n   = colBlk + nl;
            const int kw0 = (kt >> 3) + kh * 8;          // first word-row for this thread
            const int* qcol = qweight + (size_t)kw0 * N + n;
            unsigned w[8];
            #pragma unroll
            for (int i = 0; i < 8; ++i)
                w[i] = (unsigned)qcol[(size_t)i * N];    // batched column loads
            unsigned zw = (unsigned)qzeros[(size_t)g * Nw + (n >> 3)];
            int   z = (int)((zw >> ((n & 7) * 4)) & 15u) + 1;   // GPTQ +1
            float s = scales[(size_t)g * N + n];
            _Float16 sh = (_Float16)s;
            _Float16 ch = (_Float16)(-(1024.0f + (float)z) * s);
            v2h s2 = {sh, sh};
            v2h c2 = {ch, ch};
            _Float16* brow = &Bs[nl * LDS_STRIDE + kh * 64];
            #pragma unroll
            for (int i = 0; i < 8; ++i) {
                v8h b;
                #pragma unroll
                for (int p = 0; p < 4; ++p) {
                    unsigned t  = w[i] >> (8 * p);
                    unsigned pk = (t & 0xFu) | ((t & 0xF0u) << 12) | 0x64006400u;
                    v2h q2 = __builtin_bit_cast(v2h, pk);        // {1024+q0, 1024+q1}
                    v2h r2 = q2 * s2 + c2;                       // v_pk_fma_f16
                    b[2 * p]     = r2[0];
                    b[2 * p + 1] = r2[1];
                }
                *(v8h*)&brow[i << 3] = b;                // 16B-aligned b128 stores
            }
        }

        // speculative prefetch of next A tile
        if (kt + BK < K) {
            const float* nA = Ablk + BK + (size_t)(tid & 127) * K + ((tid >> 7) << 4);
            __builtin_prefetch(nA, 0, 0);
        }

        __syncthreads();

        // ---------- compute: per kk, 2 A-frags + 4 B-frags feed 8 WMMAs ----------
        const _Float16* abase = &As[(wm * 32 + lr) * LDS_STRIDE + hsel * 8];
        const _Float16* bbase = &Bs[(wn * 64 + lr) * LDS_STRIDE + hsel * 8];
        #pragma unroll
        for (int kk = 0; kk < BK; kk += 32) {
            v16h af[2];
            #pragma unroll
            for (int i = 0; i < 2; ++i) {
                const _Float16* ap = abase + i * 16 * LDS_STRIDE + kk;
                v8h alo = *(const v8h*)ap;
                v8h ahi = *(const v8h*)(ap + 16);
                af[i] = __builtin_shufflevector(alo, ahi,
                            0,1,2,3,4,5,6,7,8,9,10,11,12,13,14,15);
            }
            v16h bf[4];
            #pragma unroll
            for (int j = 0; j < 4; ++j) {
                const _Float16* bp = bbase + j * 16 * LDS_STRIDE + kk;
                v8h blo = *(const v8h*)bp;
                v8h bhi = *(const v8h*)(bp + 16);
                bf[j] = __builtin_shufflevector(blo, bhi,
                            0,1,2,3,4,5,6,7,8,9,10,11,12,13,14,15);
            }
            #pragma unroll
            for (int i = 0; i < 2; ++i)
                #pragma unroll
                for (int j = 0; j < 4; ++j)
                    acc[i][j] = __builtin_amdgcn_wmma_f32_16x16x32_f16(
                                    false, af[i], false, bf[j],
                                    (short)0, acc[i][j], false, false);
        }
    }

    // ---------- store: 8 tiles of 16x16 per wave, nontemporal ----------
    #pragma unroll
    for (int i = 0; i < 2; ++i) {
        #pragma unroll
        for (int j = 0; j < 4; ++j) {
            float* c0 = out + (size_t)(rowBlk + wm * 32 + i * 16) * N
                            + colBlk + wn * 64 + j * 16 + lr;
            #pragma unroll
            for (int r = 0; r < 8; ++r) {
                int m = r + hsel * 8;
                __builtin_nontemporal_store(acc[i][j][r], c0 + (size_t)m * N);
            }
        }
    }
}

extern "C" void kernel_launch(void* const* d_in, const int* in_sizes, int n_in,
                              void* d_out, int out_size, void* d_ws, size_t ws_size,
                              hipStream_t stream) {
    const float* x      = (const float*)d_in[0];
    const int*   qw     = (const int*)  d_in[1];
    const float* scales = (const float*)d_in[2];
    const int*   qzeros = (const int*)  d_in[3];
    // d_in[4] = g_idx: standard grouping (k / 128), folded into kernel arithmetic.

    const int K = in_sizes[4];                                   // 4096
    const int M = in_sizes[0] / K;                               // 4096
    const int N = (int)(((long long)in_sizes[1] * 8) / K);       // 11008

    dim3 grid((unsigned)(N / BN), (unsigned)(M / BM));
    dim3 block(256);
    gptq_wmma_kernel<<<grid, block, 0, stream>>>(x, qw, scales, qzeros,
                                                 (float*)d_out, M, N, K);
}
